// GRACE_76716705841326
// MI455X (gfx1250) — matrix operand order repaired and checked by reading, not compiled, not run on hardware
//
#include <hip/hip_runtime.h>
#include <hip/hip_bf16.h>

// ---------------------------------------------------------------------------
// Problem constants: B=32, L=2048, D=1024, M=256, E=128, S=128 -> N_nodes=512,
// H=2048, P=512.  Mtot = 32*512 = 16384 rows.
// ---------------------------------------------------------------------------
#define B_    32
#define L_    2048
#define D_    1024
#define M_    256
#define E_    128
#define S_    128
#define NNODE 512
#define H_    2048
#define P_    512
#define MTOT  (B_ * NNODE)   // 16384

typedef __attribute__((ext_vector_type(16))) __bf16 v16bf;
typedef __attribute__((ext_vector_type(8)))  float  v8f;
typedef __attribute__((ext_vector_type(4)))  unsigned int v4u;
typedef __attribute__((ext_vector_type(8)))  int    v8i;

__device__ __forceinline__ unsigned short f32_to_bf16(float f) {
  unsigned int u = __float_as_uint(f);
  u += 0x7FFFu + ((u >> 16) & 1u);   // round-to-nearest-even
  return (unsigned short)(u >> 16);
}

// ---------------------------------------------------------------------------
// TDM descriptor helpers (CDNA5 ISA ch.8).  data_size = 3 (8-byte units).
// group0: [1:0]=count=1 | [63:32]=lds_addr | [120:64]=global_addr | [127:126]=2
// group1: [17:16]=data_size | [79:48]=tensor_dim0 | [111:80]=tensor_dim1 |
//         [127:112]=tile_dim0 | [143:128]=tile_dim1 | [207:160]=dim0_stride
// ---------------------------------------------------------------------------
__device__ __forceinline__ v4u tdm_g0(unsigned lds_addr, unsigned long long ga) {
  v4u g;
  g[0] = 1u;                                            // count=1, user desc
  g[1] = lds_addr;
  g[2] = (unsigned)ga;
  g[3] = (unsigned)((ga >> 32) & 0x1FFFFFFu) | (2u << 30);  // addr[56:32], type=2
  return g;
}

__device__ __forceinline__ v8i tdm_g1(unsigned dim0, unsigned dim1,
                                      unsigned tile0, unsigned tile1,
                                      unsigned long long stride0) {
  v8i g;
  g[0] = (int)(3u << 16);                                   // data_size = 8B
  g[1] = (int)((dim0 & 0xFFFFu) << 16);                     // dim0[15:0] -> [31:16]
  g[2] = (int)((dim0 >> 16) | ((dim1 & 0xFFFFu) << 16));    // dim0[31:16], dim1[15:0]
  g[3] = (int)((dim1 >> 16) | ((tile0 & 0xFFFFu) << 16));   // dim1[31:16], tile_dim0
  g[4] = (int)(tile1 & 0xFFFFu);                            // tile_dim1 (tile_dim2=0)
  g[5] = (int)(stride0 & 0xFFFFFFFFull);                    // dim0_stride[31:0]
  g[6] = (int)((stride0 >> 32) & 0xFFFFull);                // dim0_stride[47:32]
  g[7] = 0;
  return g;
}

__device__ __forceinline__ void tdm_load(v4u g0, v8i g1) {
  // 2-group form: VADDR2/VADDR3 = NULL (tensors up to 2D). Portable inline-asm
  // path (bypasses the toolchain-dependent builtin arity).
  asm volatile("tensor_load_to_lds %0, %1" : : "s"(g0), "s"(g1) : "memory");
}

// ---------------------------------------------------------------------------
// 1) Gather mention / sentence rows into nodes[b][0:256] and nodes[b][384:512]
// ---------------------------------------------------------------------------
__global__ void gather_nodes(const float* __restrict__ seq,
                             const int* __restrict__ mpos,
                             const int* __restrict__ spos,
                             float* __restrict__ nodes) {
  int blk = blockIdx.x;
  int b = blk / (M_ + S_);
  int r = blk % (M_ + S_);
  int pos, dst;
  if (r < M_) { pos = mpos[b * M_ + r] + 1;        dst = r; }
  else        { pos = spos[b * S_ + (r - M_)] + 1; dst = M_ + E_ + (r - M_); }
  const float4* src  = (const float4*)(seq   + ((size_t)b * L_ + pos) * D_);
  float4*       dstp = (float4*)      (nodes + ((size_t)b * NNODE + dst) * D_);
  dstp[threadIdx.x] = src[threadIdx.x];   // 256 thr * float4 = 1024 floats
}

// ---------------------------------------------------------------------------
// 2) Segment logsumexp: one block per (b, e); mention2entity is sorted.
// ---------------------------------------------------------------------------
__global__ void entity_lse(float* __restrict__ nodes,
                           const int* __restrict__ m2e) {
  int b = blockIdx.x >> 7;          // / E_
  int e = blockIdx.x & (E_ - 1);
  __shared__ int seg[M_];
  int t = threadIdx.x;
  seg[t] = m2e[b * M_ + t];
  __syncthreads();
  int start = 0, endi = 0;
  for (int m = 0; m < M_; ++m) { int s = seg[m]; start += (s < e); endi += (s <= e); }
  int cnt = endi - start;
  const float* base = nodes + (size_t)b * NNODE * D_;
  float* dst = nodes + ((size_t)b * NNODE + M_ + e) * D_;
  for (int q = 0; q < 4; ++q) {
    int d = t + q * 256;
    float mx = -INFINITY;
    for (int m = start; m < endi; ++m)
      mx = fmaxf(mx, base[(size_t)m * D_ + d]);
    float msafe = (cnt > 0) ? mx : 0.0f;
    float s = 0.0f;
    for (int m = start; m < endi; ++m)
      s += expf(base[(size_t)m * D_ + d] - msafe);
    dst[d] = (cnt > 0) ? (logf(s) + msafe) : 0.0f;
  }
}

// ---------------------------------------------------------------------------
// Fragment-packed layouts (bf16), 1KB per 16x32(A)/32x16(B) fragment:
//   A-pack: [mt][kt][lane][slot]  m = lane&15, k = (slot<8?slot:slot+8)+8*(lane>>4)
//   B-pack: [kt][nt][lane][slot]  n = lane&15, k = (slot<8?slot:slot+8)+8*(lane>>4)
// ---------------------------------------------------------------------------
__global__ void pack_a_f32(const float* __restrict__ src,
                           unsigned short* __restrict__ dst, int KT) {
  long p = (long)blockIdx.x * 256 + threadIdx.x;
  int  slot = (int)(p & 15);
  int  lane = (int)((p >> 4) & 31);
  long rest = p >> 9;
  int  kt   = (int)(rest % KT);
  long mt   = rest / KT;
  int  K    = KT * 32;
  int  m_local = lane & 15, hi = lane >> 4;
  int  k_local = (slot < 8 ? slot : slot + 8) + 8 * hi;
  float v = src[(size_t)(mt * 16 + m_local) * K + kt * 32 + k_local];
  dst[p] = f32_to_bf16(v);
}

__global__ void pack_a_bf16(const unsigned short* __restrict__ src,
                            unsigned short* __restrict__ dst, int KT) {
  long p = (long)blockIdx.x * 256 + threadIdx.x;
  int  slot = (int)(p & 15);
  int  lane = (int)((p >> 4) & 31);
  long rest = p >> 9;
  int  kt   = (int)(rest % KT);
  long mt   = rest / KT;
  int  K    = KT * 32;
  int  m_local = lane & 15, hi = lane >> 4;
  int  k_local = (slot < 8 ? slot : slot + 8) + 8 * hi;
  dst[p] = src[(size_t)(mt * 16 + m_local) * K + kt * 32 + k_local];
}

__global__ void pack_w(const float* __restrict__ w,
                       unsigned short* __restrict__ dst, int NT) {
  long p = (long)blockIdx.x * 256 + threadIdx.x;
  int  slot = (int)(p & 15);
  int  lane = (int)((p >> 4) & 31);
  long rest = p >> 9;
  int  nt   = (int)(rest % NT);
  long kt   = rest / NT;
  int  N    = NT * 16;
  int  n_local = lane & 15, khalf = lane >> 4;
  int  k_local = (slot < 8 ? slot : slot + 8) + 8 * khalf;
  float v = w[(size_t)(kt * 32 + k_local) * N + nt * 16 + n_local];
  dst[p] = f32_to_bf16(v);
}

// ---------------------------------------------------------------------------
// 3) GEMM with TDM->LDS double buffering.
//    Block = 256 thr = 8 waves (2x4). Wave tile 64x64 = 4x4 WMMA tiles.
//    Per K-step (BK=32): wave0 issues 2 tensor_load_to_lds (A: 2D 8x1KB rows
//    stride KT KB; B: 1D 16KB), everyone consumes via ds_load_b128, 16 WMMA.
//    Stage = 24KB, double buffered = 48KB LDS of the 320KB WGP pool.
// ---------------------------------------------------------------------------
template <bool STORE_BF16>
__global__ __launch_bounds__(256)
void gemm_wmma_bf16(const unsigned short* __restrict__ Apk,
                    const unsigned short* __restrict__ Bpk,
                    const float* __restrict__ bias, void* __restrict__ outp,
                    int KT, int NT, int Ncols) {
  __shared__ __align__(1024) char smem[2 * 24576];

  const int lane = threadIdx.x & 31;
  const int wave = threadIdx.x >> 5;
  const int wm = wave >> 2, wn = wave & 3;

  const unsigned ldsbase = (unsigned)(size_t)(void*)smem;
  const unsigned long long gA0 =
      (unsigned long long)(size_t)Apk + (size_t)(blockIdx.x * 8) * KT * 1024ull;
  const unsigned long long gB0 =
      (unsigned long long)(size_t)Bpk + (size_t)(blockIdx.y * 16) * 1024ull;

  // kt-invariant group1 descriptors (units of 8 bytes)
  const v8i g1A = tdm_g1(/*dim0*/128, /*dim1*/8, /*tile0*/128, /*tile1*/8,
                         /*stride0*/(unsigned long long)KT * 128ull);
  const v8i g1B = tdm_g1(2048, 1, 2048, 1, 2048ull);

  // Force a scalar branch: TDM ignores EXEC, so only wave 0 may reach the asm.
  const bool producer = (__builtin_amdgcn_readfirstlane((int)threadIdx.x) < 32);

  const v8f vzero = {0.f, 0.f, 0.f, 0.f, 0.f, 0.f, 0.f, 0.f};
  v8f acc[4][4];
#pragma unroll
  for (int i = 0; i < 4; ++i)
#pragma unroll
    for (int j = 0; j < 4; ++j) acc[i][j] = vzero;

  if (producer) {
    tdm_load(tdm_g0(ldsbase, gA0), g1A);
    tdm_load(tdm_g0(ldsbase + 8192, gB0 /* kt=0 */), g1B);
  }

  for (int kt = 0; kt < KT; ++kt) {
    const int buf = kt & 1;
    if (producer) __builtin_amdgcn_s_wait_tensorcnt(0);
    __syncthreads();   // stage `buf` ready; previous compute on buf^1 done
    if (producer && (kt + 1 < KT)) {
      unsigned stage = ldsbase + (buf ^ 1) * 24576;
      tdm_load(tdm_g0(stage, gA0 + (unsigned long long)(kt + 1) * 1024ull), g1A);
      tdm_load(tdm_g0(stage + 8192,
                      gB0 + (unsigned long long)(kt + 1) * (unsigned long long)NT * 1024ull),
               g1B);
    }

    const char* stage = smem + buf * 24576;
    v16bf a[4], b[4];
#pragma unroll
    for (int i = 0; i < 4; ++i)
      a[i] = ((const v16bf*)(stage + (wm * 4 + i) * 1024))[lane];
#pragma unroll
    for (int j = 0; j < 4; ++j)
      b[j] = ((const v16bf*)(stage + 8192 + (wn * 4 + j) * 1024))[lane];
#pragma unroll
    for (int i = 0; i < 4; ++i)
#pragma unroll
      for (int j = 0; j < 4; ++j)
        acc[i][j] = __builtin_amdgcn_wmma_f32_16x16x32_bf16(
            false, a[i], false, b[j], (short)0, acc[i][j], false, false);
  }

  // C/D layout: VGPR r -> M = r (+8 for lanes 16..31), N = lane & 15
  const int row0 = blockIdx.x * 128 + wm * 64 + ((lane >> 4) << 3);
  const int col0 = blockIdx.y * 256 + wn * 64 + (lane & 15);
#pragma unroll
  for (int i = 0; i < 4; ++i)
#pragma unroll
    for (int j = 0; j < 4; ++j) {
      int col = col0 + j * 16;
      float bv = bias[col];
#pragma unroll
      for (int r = 0; r < 8; ++r) {
        int row = row0 + i * 16 + r;
        float v = acc[i][j][r] + bv;
        if (STORE_BF16)
          ((unsigned short*)outp)[(size_t)row * Ncols + col] = f32_to_bf16(v);
        else
          ((float*)outp)[(size_t)row * Ncols + col] = v;
      }
    }
}

// ---------------------------------------------------------------------------
// Workspace layout (bytes):
//   [0,       64M)  nodes f32            (later reused as A2-packed bf16)
//   [64M,     96M)  A1-packed bf16       (16384 x 1024)
//   [96M,    100M)  W1-packed bf16       (1024 x 2048)
//   [100M,   102M)  W2-packed bf16       (2048 x 512)
//   [102M,   166M)  h bf16 row-major     (16384 x 2048)
// ---------------------------------------------------------------------------
extern "C" void kernel_launch(void* const* d_in, const int* in_sizes, int n_in,
                              void* d_out, int out_size, void* d_ws, size_t ws_size,
                              hipStream_t stream) {
  const float* seq    = (const float*)d_in[0];
  const int*   mpos   = (const int*)  d_in[1];
  const int*   m2e    = (const int*)  d_in[2];
  const int*   spos   = (const int*)  d_in[3];
  const float* fc1_w  = (const float*)d_in[4];
  const float* fc1_b  = (const float*)d_in[5];
  const float* fc2_w  = (const float*)d_in[6];
  const float* fc2_b  = (const float*)d_in[7];
  float*       out    = (float*)d_out;

  char* ws = (char*)d_ws;
  const size_t MB = 1024ull * 1024ull;
  float*          nodes = (float*)         (ws + 0);
  unsigned short* A1    = (unsigned short*)(ws + 64 * MB);
  unsigned short* W1p   = (unsigned short*)(ws + 96 * MB);
  unsigned short* W2p   = (unsigned short*)(ws + 100 * MB);
  unsigned short* Hb    = (unsigned short*)(ws + 102 * MB);
  unsigned short* A2    = (unsigned short*)(ws + 0);   // aliases nodes (dead by then)

  // 1) gather
  gather_nodes<<<B_ * (M_ + S_), 256, 0, stream>>>(seq, mpos, spos, nodes);
  // 2) entity logsumexp
  entity_lse<<<B_ * E_, 256, 0, stream>>>(nodes, m2e);
  // 3) pack activations + weights into WMMA fragment order (bf16)
  pack_a_f32<<<(MTOT * D_) / 256, 256, 0, stream>>>(nodes, A1, D_ / 32);
  pack_w<<<(D_ * H_) / 256, 256, 0, stream>>>(fc1_w, W1p, H_ / 16);
  pack_w<<<(H_ * P_) / 256, 256, 0, stream>>>(fc2_w, W2p, P_ / 16);
  // 4) GEMM1: h = nodes @ fc1_w + fc1_b  -> bf16 row-major
  gemm_wmma_bf16<true><<<dim3(MTOT / 128, H_ / 256), 256, 0, stream>>>(
      A1, W1p, fc1_b, Hb, D_ / 32, H_ / 16, H_);
  // 5) repack h into A-fragment order for GEMM2
  pack_a_bf16<<<(MTOT * H_) / 256, 256, 0, stream>>>(Hb, A2, H_ / 32);
  // 6) GEMM2: out = h @ fc2_w + fc2_b -> f32
  gemm_wmma_bf16<false><<<dim3(MTOT / 128, P_ / 256), 256, 0, stream>>>(
      A2, W2p, fc2_b, out, H_ / 32, P_ / 16, P_);
}
